// SampleRNN_74826920231302
// MI455X (gfx1250) — matrix operand order repaired and checked by reading, not compiled, not run on hardware
//
// CDNA5 / gfx1250 GRU sample-RNN forward.
//
// proj-table precompute turns the input GEMM into a gather; 16 independent
// batch-slice workgroups run the 4096-step recurrence with bf16 WMMA
// (16x16x32) gate GEMMs and register-resident B fragments; logits are a
// separate whole-GPU WMMA GEMM at the HBM write floor (A fragments register-
// resident across all 16 N-tiles, W_cls staged in LDS) when ws_size permits
// storing H (bf16, 256 MB), else fused into the scan.

#include <hip/hip_runtime.h>
#include <stdint.h>

#define BS   256
#define SEQ  4096
#define D    128
#define NC   256

typedef __attribute__((ext_vector_type(16))) __bf16 v16bf;
typedef __attribute__((ext_vector_type(8)))  float  v8f;

union FragU { uint4 q[2]; v16bf v; };

__device__ __forceinline__ v8f wmma_bf16(v16bf a, v16bf b, v8f c) {
  // D = A(16x32 bf16) * B(32x16 bf16) + C(16x16 f32)
  return __builtin_amdgcn_wmma_f32_16x16x32_bf16(false, a, false, b, (short)0, c,
                                                 false, false);
}

// Build a B fragment from column n of W (p points at K=0 of this k-step, col n).
// Dense 32x16 bf16 B layout: lane <-> N (n = lane&15); element e holds
// K = 8*half + e (e<8), K = 16 + 8*half + (e-8) (e>=8).
__device__ __forceinline__ v16bf load_b_frag_f32(const float* __restrict__ p, int ld, int half) {
  v16bf r;
#pragma unroll
  for (int e = 0; e < 16; ++e) r[e] = (__bf16)p[((e < 8) ? (8 * half + e) : (8 + 8 * half + e)) * ld];
  return r;
}

// Column gather of bf16 (stride ld) -> B fragment, same K order as above.
__device__ __forceinline__ v16bf load_b_frag_bf16(const __bf16* p, int ld, int half) {
  v16bf r;
#pragma unroll
  for (int e = 0; e < 16; ++e) r[e] = p[((e < 8) ? (8 * half + e) : (8 + 8 * half + e)) * ld];
  return r;
}

// A fragment (16x32 bf16). Per ISA: lane holds row (lane&15); half=lane>>4;
// VGPR0-3: K = 8*half + {0..7}; VGPR4-7: K = 16 + 8*half + {0..7}.
// rowbase points at row r, K=0 (bf16, contiguous K). Offsets are 16B-aligned.
__device__ __forceinline__ v16bf load_a_frag(const char* rowbase, int kb, int half) {
  FragU u;
  const uint4* p = (const uint4*)(rowbase + kb * 64 + half * 16);
  u.q[0] = p[0];
  u.q[1] = p[2];   // +32 bytes
  return u.v;
}

__device__ __forceinline__ float sigm(float v) { return 1.0f / (1.0f + __expf(-v)); }

// ---------------------------------------------------------------------------
// Kernel 1: proj[c][j] = embed[c] . Wi[:,j] + bi[j]   (257 x 384, tiny)
// ---------------------------------------------------------------------------
__global__ __launch_bounds__(256) void proj_embed(
    const float* __restrict__ embed, const float* __restrict__ Wi,
    const float* __restrict__ bi, float* __restrict__ proj) {
  int idx = blockIdx.x * 256 + threadIdx.x;
  if (idx >= (NC + 1) * 3 * D) return;
  int c = idx / (3 * D), j = idx % (3 * D);
  float acc = bi[j];
  const float* e = embed + c * D;
#pragma unroll 4
  for (int k = 0; k < D; ++k) acc += e[k] * Wi[k * (3 * D) + j];
  proj[idx] = acc;
}

// Kernel 1b: W_cls -> bf16 (row-major, for the split logits GEMM)
__global__ __launch_bounds__(256) void wcls_to_bf16(const float* __restrict__ W,
                                                    __bf16* __restrict__ Wb) {
  int idx = blockIdx.x * 256 + threadIdx.x;  // D*NC = 32768 exact
  Wb[idx] = (__bf16)W[idx];
}

// ---------------------------------------------------------------------------
// Kernel 2: the serial GRU scan. grid = 16 WGs x 256 threads (8 waves).
// WG owns batch rows b0..b0+15; wave w owns hidden cols 16w..16w+15 of each
// gate, so r/z/n align in its own C fragments (no cross-wave exchange).
// FUSED: also computes logits(h_{t-1}) in the same WMMA pass (A shared).
// ---------------------------------------------------------------------------
template <bool FUSED>
__global__ __launch_bounds__(256) void gru_scan(
    const int* __restrict__ x,
    const float* __restrict__ Wh_rz, const float* __restrict__ Wh_n,
    const float* __restrict__ bh_n,
    const float* __restrict__ W_cls, const float* __restrict__ b_cls,
    const float* __restrict__ init_state,
    const float* __restrict__ proj,    // (NC+1, 3D) f32
    float* __restrict__ out,           // (BS, SEQ, NC) f32 (FUSED)
    __bf16* __restrict__ Hbuf)         // (SEQ, BS, D) bf16 (!FUSED)
{
  constexpr int HSA  = 136;  // bf16 row stride (272B, padded: ~2-way banks)
  constexpr int HS32 = 132;  // f32 row stride (padded)
  __shared__ __align__(16) __bf16 sHA[2][16 * HSA];   // h, WMMA-A feed
  __shared__ float sH32[2][16 * HS32];                // h, f32 master
  __shared__ int   sTok[2][16];

  const int tid  = threadIdx.x;
  const int lane = tid & 31;
  const int w    = tid >> 5;         // wave 0..7
  const int half = (lane >> 4) & 1;
  const int ln   = lane & 15;
  const int jcol = 16 * w + ln;      // hidden column handled by this lane
  const int b0   = blockIdx.x * 16;

  // init h = init_state broadcast
#pragma unroll
  for (int i = 0; i < 8; ++i) {
    int e = tid + 256 * i;           // 0..2047
    int m = e >> 7, j = e & 127;
    float v = init_state[j];
    sH32[0][m * HS32 + j] = v;
    sHA[0][m * HSA + j]   = (__bf16)v;
  }

  // Hoist all B fragments into registers (loop-invariant).
  v16bf Br[4], Bz[4], Bn[4], Bl0[4], Bl1[4];
#pragma unroll
  for (int kb = 0; kb < 4; ++kb) {
    int k = 32 * kb;                 // K base of this step; lane <-> N column
    Br[kb] = load_b_frag_f32(Wh_rz + k * 256 + jcol, 256, half);
    Bz[kb] = load_b_frag_f32(Wh_rz + k * 256 + 128 + jcol, 256, half);
    Bn[kb] = load_b_frag_f32(Wh_n  + k * 128 + jcol, 128, half);
    if (FUSED) {
      Bl0[kb] = load_b_frag_f32(W_cls + k * 256 + jcol, 256, half);
      Bl1[kb] = load_b_frag_f32(W_cls + k * 256 + 128 + jcol, 256, half);
    }
  }
  const float bhnj = bh_n[jcol];
  float bc0 = 0.f, bc1 = 0.f;
  if (FUSED) { bc0 = b_cls[jcol]; bc1 = b_cls[128 + jcol]; }
  __syncthreads();

  const int tmax = FUSED ? SEQ : (SEQ - 1);
  for (int t = 0; t <= tmax; ++t) {
    const int p = t & 1;

    // Gather xi rows from proj (L2 resident). Issued before WMMA chain.
    float xr[8], xz[8], xn[8];
#pragma unroll
    for (int i = 0; i < 8; ++i) {
      int m = i + 8 * half;
      int tok = (t == 0) ? NC : sTok[p][m];
      const float* pp = proj + (size_t)tok * (3 * D) + jcol;
      xr[i] = pp[0];
      xz[i] = pp[D];
      xn[i] = pp[2 * D];
    }

    // A fragments of h_{t-1}
    const char* arow = (const char*)&sHA[p][0] + ln * (HSA * 2);
    v16bf a[4];
#pragma unroll
    for (int kb = 0; kb < 4; ++kb) a[kb] = load_a_frag(arow, kb, half);

    v8f cr = {}, cz = {}, cn = {}, cl0 = {}, cl1 = {};
#pragma unroll
    for (int kb = 0; kb < 4; ++kb) {
      cr = wmma_bf16(a[kb], Br[kb], cr);
      cz = wmma_bf16(a[kb], Bz[kb], cz);
      cn = wmma_bf16(a[kb], Bn[kb], cn);
      if (FUSED) {
        cl0 = wmma_bf16(a[kb], Bl0[kb], cl0);
        cl1 = wmma_bf16(a[kb], Bl1[kb], cl1);
      }
    }

    // Prefetch next step's tokens (into the other buffer; no race).
    if (tid < 16 && t < SEQ) sTok[1 - p][tid] = x[(size_t)(b0 + tid) * SEQ + t];

    // Elementwise GRU cell; write h_t into the other LDS buffer.
    if (!FUSED || t < SEQ) {
#pragma unroll
      for (int i = 0; i < 8; ++i) {
        int m = i + 8 * half;
        float hprev = sH32[p][m * HS32 + jcol];
        float r  = sigm(xr[i] + cr[i]);
        float z  = sigm(xz[i] + cz[i]);
        float nn = tanhf(xn[i] + r * (cn[i] + bhnj));
        float hnew = (1.0f - z) * nn + z * hprev;
        sH32[1 - p][m * HS32 + jcol] = hnew;
        sHA[1 - p][m * HSA + jcol]   = (__bf16)hnew;
        if (!FUSED)
          Hbuf[((size_t)t * BS + (b0 + m)) * D + jcol] = (__bf16)hnew;
      }
    }

    // Logits of h_{t-1} (fused path); fire-and-forget stores.
    if (FUSED && t > 0) {
      const size_t trow = (size_t)(t - 1) * NC;
#pragma unroll
      for (int i = 0; i < 8; ++i) {
        int m = i + 8 * half;
        size_t base = (size_t)(b0 + m) * ((size_t)SEQ * NC) + trow;
        out[base + jcol]       = cl0[i] + bc0;
        out[base + 128 + jcol] = cl1[i] + bc1;
      }
    }
    __syncthreads();
  }
}

// ---------------------------------------------------------------------------
// Kernel 3 (split path): logits = H @ W_cls + b_cls over all (t,b).
// One M-tile (16 H rows) per wave; A fragments stay in registers while the
// wave sweeps ALL 16 N-tiles (64 WMMAs). W_cls (bf16) staged once per WG in
// LDS with a padded stride. HBM traffic = 256 MB H read + 1 GB logits write.
// ---------------------------------------------------------------------------
__global__ __launch_bounds__(256) void gru_logits(
    const __bf16* __restrict__ Hbuf,   // (SEQ*BS, D) bf16
    const __bf16* __restrict__ Wc16,   // (D, NC) bf16
    const float* __restrict__ b_cls,
    float* __restrict__ out) {
  constexpr int WSTR = NC + 8;         // 264 bf16 = 528 B row stride (16B mult)
  __shared__ __align__(16) __bf16 sW[D * WSTR];   // ~67.6 KB
  __shared__ float sBc[NC];

  const int tid  = threadIdx.x;
  const int lane = tid & 31;
  const int wv   = tid >> 5;
  const int half = (lane >> 4) & 1;
  const int ln   = lane & 15;

  // Stage W_cls bf16 into LDS (dword-wise), plus the bias.
  {
    const uint32_t* src = (const uint32_t*)Wc16;      // D*NC/2 = 16384 dwords
    uint32_t* dst = (uint32_t*)sW;
#pragma unroll
    for (int i = 0; i < (D * NC / 2) / 256; ++i) {    // 64 iters
      int e = tid + 256 * i;
      int k = e / (NC / 2), c2 = e % (NC / 2);
      dst[k * (WSTR / 2) + c2] = src[e];
    }
    if (tid < NC) sBc[tid] = b_cls[tid];
  }
  __syncthreads();

  // This wave's M-tile: 16 flat rows (t*BS + b) of H.
  const size_t m0 = ((size_t)blockIdx.x * 8 + wv) * 16;
  const char* arow = (const char*)(Hbuf + (m0 + (size_t)ln) * D);
  v16bf a[4];
#pragma unroll
  for (int kb = 0; kb < 4; ++kb) a[kb] = load_a_frag(arow, kb, half);

  // Per-element output addresses (row-dependent part hoisted).
  size_t obase[8];
#pragma unroll
  for (int i = 0; i < 8; ++i) {
    size_t m  = m0 + i + 8 * half;    // flat row = t*BS + b
    size_t tt = m / BS;
    size_t b  = m % BS;
    obase[i] = (b * SEQ + tt) * NC + (size_t)ln;
  }

#pragma unroll
  for (int nt = 0; nt < NC / 16; ++nt) {
    v8f c = {};
#pragma unroll
    for (int kb = 0; kb < 4; ++kb) {
      v16bf bf = load_b_frag_bf16(sW + (32 * kb) * WSTR + nt * 16 + ln, WSTR, half);
      c = wmma_bf16(a[kb], bf, c);
    }
    const float bc = sBc[nt * 16 + ln];
#pragma unroll
    for (int i = 0; i < 8; ++i) out[obase[i] + nt * 16] = c[i] + bc;
  }
}

// ---------------------------------------------------------------------------
extern "C" void kernel_launch(void* const* d_in, const int* in_sizes, int n_in,
                              void* d_out, int out_size, void* d_ws, size_t ws_size,
                              hipStream_t stream) {
  (void)in_sizes; (void)n_in; (void)out_size;
  const int*   x     = (const int*)d_in[0];
  const float* embed = (const float*)d_in[1];
  const float* Wi    = (const float*)d_in[2];
  const float* bi    = (const float*)d_in[3];
  const float* Whrz  = (const float*)d_in[4];
  const float* Whn   = (const float*)d_in[5];
  const float* bhn   = (const float*)d_in[6];
  const float* Wcls  = (const float*)d_in[7];
  const float* bcls  = (const float*)d_in[8];
  const float* h0    = (const float*)d_in[9];
  float* out = (float*)d_out;

  const size_t PROJ_BYTES = (size_t)(NC + 1) * 3 * D * 4;   // 394,752
  const size_t WC_BYTES   = (size_t)D * NC * 2;             // 65,536
  const size_t H_BYTES    = (size_t)SEQ * BS * D * 2;       // 256 MiB
  char* wsb = (char*)d_ws;
  float*  proj = (float*)wsb;
  size_t off1 = (PROJ_BYTES + 255) & ~(size_t)255;
  __bf16* Wc16 = (__bf16*)(wsb + off1);
  size_t off2 = (off1 + WC_BYTES + 255) & ~(size_t)255;
  __bf16* Hbuf = (__bf16*)(wsb + off2);
  const bool split = (ws_size >= off2 + H_BYTES);  // deterministic per-run

  const int projN = (NC + 1) * 3 * D;
  proj_embed<<<(projN + 255) / 256, 256, 0, stream>>>(embed, Wi, bi, proj);

  if (split) {
    wcls_to_bf16<<<(D * NC) / 256, 256, 0, stream>>>(Wcls, Wc16);
    gru_scan<false><<<BS / 16, 256, 0, stream>>>(x, Whrz, Whn, bhn, Wcls, bcls,
                                                 h0, proj, out, Hbuf);
    const int nblk = (SEQ * BS / 16) / 8;                   // 8192 blocks
    gru_logits<<<nblk, 256, 0, stream>>>(Hbuf, Wc16, bcls, out);
  } else {
    gru_scan<true><<<BS / 16, 256, 0, stream>>>(x, Whrz, Whn, bhn, Wcls, bcls,
                                                h0, proj, out, nullptr);
  }
}
